// EncodeProcessDecode_22024592294283
// MI455X (gfx1250) — matrix operand order repaired and checked by reading, not compiled, not run on hardware
//
#include <hip/hip_runtime.h>

// ---------------------------------------------------------------------------
// EncodeProcessDecode GNN for MI455X (gfx1250), f16 WMMA path, v2.
//   - V_WMMA_F32_16X16X32_F16; per-wave register blocking: 64(M) x 16(N)
//     tile = 4 accumulators sharing one B fragment per K-step
//     (2.5 b128 loads / WMMA instead of 8).
//   - Block = 256 threads = 8 waves = 64 x 128 output tile; LayerNorm
//     epilogue in-block via LDS (full LATENT=128 row per block).
//   - Activation buffers feeding GEMM-A are padded to 64-row multiples;
//     pad rows compute garbage that is never stored (stores guarded).
//   - Residual streams h,e: fp32 masters + f16 mirrors. segment_sum via
//     global atomicAdd(float).
// ---------------------------------------------------------------------------

#define N_NODES 10000
#define N_EDGES 100000
#define LATENT  128
#define MP_N    10048    // N_NODES padded to 64
#define MP_E    100032   // N_EDGES padded to 64

typedef __attribute__((ext_vector_type(16))) _Float16 v16h;
typedef __attribute__((ext_vector_type(8)))  _Float16 v8h;
typedef __attribute__((ext_vector_type(8)))  float    v8f;

enum { F_RELU = 1, F_LN = 2 };

// Pack W[K][N] fp32 (row-major) -> Wt[Np][Kp] f16 (transposed, zero padded).
__global__ __launch_bounds__(256) void pack_wt_kernel(
    const float* __restrict__ W, _Float16* __restrict__ Wt,
    int K, int N, int Kp, int Np)
{
    int i = blockIdx.x * 256 + threadIdx.x;
    if (i >= Np * Kp) return;
    int n = i / Kp, k = i % Kp;
    float v = (n < N && k < K) ? W[(long)k * N + n] : 0.f;
    Wt[i] = (_Float16)v;
}

// fp32 [M][C] -> f16 [M][Cp] zero-padded columns.
__global__ __launch_bounds__(256) void cvt_pad_kernel(
    const float* __restrict__ src, _Float16* __restrict__ dst,
    int M, int C, int Cp)
{
    long i = blockIdx.x * 256L + threadIdx.x;
    if (i >= (long)M * Cp) return;
    int r = (int)(i / Cp), c = (int)(i % Cp);
    dst[i] = (_Float16)((c < C) ? src[(long)r * C + c] : 0.f);
}

__global__ __launch_bounds__(256) void fill0_kernel(float* __restrict__ p, long n)
{
    long i = blockIdx.x * 256L + threadIdx.x;
    if (i < n) p[i] = 0.f;
}

// msg[e] = [ h16[recv[e]] | h16[send[e]] | e16[e] ]  (384 halves = 48 x v8h)
__global__ __launch_bounds__(256) void gather_concat_kernel(
    const _Float16* __restrict__ h16, const _Float16* __restrict__ e16,
    const int* __restrict__ recv, const int* __restrict__ send,
    _Float16* __restrict__ msg)
{
    long i = blockIdx.x * 256L + threadIdx.x;
    if (i >= (long)N_EDGES * 48) return;
    int e = (int)(i / 48), c = (int)(i % 48);
    const _Float16* src;
    if (c < 16)      src = h16 + (long)recv[e] * LATENT + c * 8;
    else if (c < 32) src = h16 + (long)send[e] * LATENT + (c - 16) * 8;
    else             src = e16 + (long)e       * LATENT + (c - 32) * 8;
    *(v8h*)(msg + (long)e * 384 + c * 8) = *(const v8h*)src;
}

// agg[recv[e]] += enew[e]  (fp32 atomics)
__global__ __launch_bounds__(256) void scatter_add_kernel(
    const float* __restrict__ enew, const int* __restrict__ recv,
    float* __restrict__ agg)
{
    long i = blockIdx.x * 256L + threadIdx.x;
    if (i >= (long)N_EDGES * LATENT) return;
    int e = (int)(i >> 7), c = (int)(i & 127);
    atomicAdd(agg + (long)recv[e] * LATENT + c, enew[i]);
}

// nin[n] = [ f16(agg[n]) | h16[n] ]  (256 halves)
__global__ __launch_bounds__(256) void concat_node_kernel(
    const float* __restrict__ agg, const _Float16* __restrict__ h16,
    _Float16* __restrict__ nin)
{
    long i = blockIdx.x * 256L + threadIdx.x;
    if (i >= (long)N_NODES * 256) return;
    int n = (int)(i >> 8), c = (int)(i & 255);
    nin[(long)n * 256 + c] = (c < 128) ? (_Float16)agg[(long)n * 128 + c]
                                       : h16[(long)n * 128 + (c - 128)];
}

// ---------------------------------------------------------------------------
// Fused GEMM + bias + (ReLU | LayerNorm) + (residual accumulate) + f16 mirror.
// Block: 64 output rows x n_tiles*16 cols; wave w owns n-tile w across all
// four 16-row sub-tiles (acc0..acc3 share one B fragment per K-step).
// A fragment (16x32 f16, ISA 7.12.2): lane (l15,hi); row = m + l15;
//   halves { hi*8..hi*8+7 } and { hi*8+16..hi*8+23 } of the K-slice.
// B fragment: Wt[nt*16+l15][k + hi*8 .. +7] and [k + hi*8 + 16 .. +23].
// C layout: lane (l15,hi) col nt*16+l15, VGPR r -> row hi*8+r.
// ---------------------------------------------------------------------------
#define SHUF16(lo, hi) __builtin_shufflevector((lo), (hi), \
    0, 1, 2, 3, 4, 5, 6, 7, 8, 9, 10, 11, 12, 13, 14, 15)

__global__ __launch_bounds__(256) void gemm_wmma_kernel(
    const _Float16* __restrict__ A, const _Float16* __restrict__ Wt,
    const float* __restrict__ bias, const float* __restrict__ gamma,
    const float* __restrict__ beta, float* __restrict__ resid,
    float* __restrict__ outf, _Float16* __restrict__ outh,
    int M, int K, int n_tiles, int N_out, int ldo, int flags)
{
    __shared__ float Ct[64][132];          // +4 pad to spread LDS banks
    __shared__ float s_mu[64], s_rs[64];

    const int lane = threadIdx.x & 31;
    const int wave = threadIdx.x >> 5;
    const int m0   = blockIdx.x * 64;
    const int l15  = lane & 15;
    const int hi   = lane >> 4;

    if (wave < n_tiles) {
        const _Float16* Br = Wt + (long)(wave * 16 + l15) * K + hi * 8;
        const _Float16* Ar = A  + (long)(m0 + l15) * K + hi * 8;
        const long rs = 16L * K;           // bytes-in-halves per 16-row subtile
        v8f acc0 = {}, acc1 = {}, acc2 = {}, acc3 = {};
        for (int k = 0; k < K; k += 32) {
            v16h bv = SHUF16(*(const v8h*)(Br + k), *(const v8h*)(Br + k + 16));
            const _Float16* a = Ar + k;
            v16h av0 = SHUF16(*(const v8h*)(a),          *(const v8h*)(a + 16));
            v16h av1 = SHUF16(*(const v8h*)(a + rs),     *(const v8h*)(a + rs + 16));
            v16h av2 = SHUF16(*(const v8h*)(a + 2 * rs), *(const v8h*)(a + 2 * rs + 16));
            v16h av3 = SHUF16(*(const v8h*)(a + 3 * rs), *(const v8h*)(a + 3 * rs + 16));
            acc0 = __builtin_amdgcn_wmma_f32_16x16x32_f16(false, av0, false, bv, (short)0, acc0, false, false);
            acc1 = __builtin_amdgcn_wmma_f32_16x16x32_f16(false, av1, false, bv, (short)0, acc1, false, false);
            acc2 = __builtin_amdgcn_wmma_f32_16x16x32_f16(false, av2, false, bv, (short)0, acc2, false, false);
            acc3 = __builtin_amdgcn_wmma_f32_16x16x32_f16(false, av3, false, bv, (short)0, acc3, false, false);
        }
        const int ncol = wave * 16 + l15;
        const int mb   = hi * 8;
#pragma unroll
        for (int r = 0; r < 8; ++r) {
            Ct[mb + r][ncol]      = acc0[r];
            Ct[16 + mb + r][ncol] = acc1[r];
            Ct[32 + mb + r][ncol] = acc2[r];
            Ct[48 + mb + r][ncol] = acc3[r];
        }
    }
    __syncthreads();

    if (flags & F_LN) {                     // only used when N_out == 128
        if (threadIdx.x < 64) {
            float s = 0.f, ss = 0.f;
            for (int n = 0; n < 128; ++n) {
                float v = Ct[threadIdx.x][n] + bias[n];
                s += v; ss += v * v;
            }
            float mu  = s * (1.f / 128.f);
            float var = ss * (1.f / 128.f) - mu * mu;
            s_mu[threadIdx.x] = mu;
            s_rs[threadIdx.x] = rsqrtf(var + 1e-5f);
        }
        __syncthreads();
    }

    const int w = n_tiles * 16;
    const int total = 64 * w;
    for (int i = threadIdx.x; i < total; i += 256) {
        int m = i / w, n = i - m * w;
        int row = m0 + m;
        if (row >= M || n >= N_out) continue;   // pad rows / pad cols: discard
        float v = Ct[m][n] + bias[n];
        if (flags & F_RELU) v = fmaxf(v, 0.f);
        if (flags & F_LN)   v = (v - s_mu[m]) * s_rs[m] * gamma[n] + beta[n];
        long o = (long)row * ldo + n;
        if (outf) outf[o] = v;              // pre-residual value (e_new path)
        float vr = v;
        if (resid) { vr = resid[o] + v; resid[o] = vr; }  // h/e master update
        if (outh) outh[o] = (_Float16)vr;   // f16 mirror feeds next GEMM
    }
}

// ---------------------------------------------------------------------------
extern "C" void kernel_launch(void* const* d_in, const int* in_sizes, int n_in,
                              void* d_out, int out_size, void* d_ws, size_t ws_size,
                              hipStream_t stream) {
    (void)in_sizes; (void)n_in; (void)out_size; (void)ws_size;

    const float* x     = (const float*)d_in[0];
    const int*   eidx  = (const int*)d_in[1];
    const float* efeat = (const float*)d_in[2];
    const int* send = eidx;             // edge_index[0] = x_j source
    const int* recv = eidx + N_EDGES;   // edge_index[1] = target / aggregation

    // Params flattened with sorted dict keys (tree_flatten order):
    // dec, enc_edge, enc_edge_ln, enc_node, enc_node_ln, gnn[ s: edge_ln,
    // edge_mlp, node_ln, node_mlp ]. Tuples (W,b)/(gamma,beta) in order.
    const float *decW[3], *decB[3], *eeW[3], *eeB[3], *enW[3], *enB[3];
    int pi = 3;
    for (int l = 0; l < 3; ++l) { decW[l] = (const float*)d_in[pi++]; decB[l] = (const float*)d_in[pi++]; }
    for (int l = 0; l < 3; ++l) { eeW[l]  = (const float*)d_in[pi++]; eeB[l]  = (const float*)d_in[pi++]; }
    const float* ee_g = (const float*)d_in[pi++]; const float* ee_b = (const float*)d_in[pi++];
    for (int l = 0; l < 3; ++l) { enW[l]  = (const float*)d_in[pi++]; enB[l]  = (const float*)d_in[pi++]; }
    const float* en_g = (const float*)d_in[pi++]; const float* en_b = (const float*)d_in[pi++];
    const float *elg[5], *elb[5], *nlg[5], *nlb[5];
    const float *eW[5][3], *eB[5][3], *nW[5][3], *nB[5][3];
    for (int s = 0; s < 5; ++s) {
        elg[s] = (const float*)d_in[pi++]; elb[s] = (const float*)d_in[pi++];
        for (int l = 0; l < 3; ++l) { eW[s][l] = (const float*)d_in[pi++]; eB[s][l] = (const float*)d_in[pi++]; }
        nlg[s] = (const float*)d_in[pi++]; nlb[s] = (const float*)d_in[pi++];
        for (int l = 0; l < 3; ++l) { nW[s][l] = (const float*)d_in[pi++]; nB[s][l] = (const float*)d_in[pi++]; }
    }

    // Workspace bump allocator (256B aligned).
    char* wsb = (char*)d_ws; size_t off = 0;
    auto alloc = [&](long bytes) -> void* {
        void* p = wsb + off; off = (off + (size_t)bytes + 255) & ~(size_t)255; return p;
    };

    // GEMM-A-feeding activation buffers use 64-row-padded row counts.
    _Float16* x16  = (_Float16*)alloc((long)MP_N * 32  * 2);
    _Float16* ef16 = (_Float16*)alloc((long)MP_E * 32  * 2);
    float*    hF   = (float*)   alloc((long)N_NODES * 128 * 4);
    _Float16* h16  = (_Float16*)alloc((long)MP_N * 128 * 2);
    float*    eF   = (float*)   alloc((long)N_EDGES * 128 * 4);
    _Float16* e16  = (_Float16*)alloc((long)MP_E * 128 * 2);
    float*    enew = (float*)   alloc((long)N_EDGES * 128 * 4);
    float*    agg  = (float*)   alloc((long)N_NODES * 128 * 4);
    _Float16* msg  = (_Float16*)alloc((long)MP_E * 384 * 2);
    _Float16* te1  = (_Float16*)alloc((long)MP_E * 128 * 2);
    _Float16* te2  = (_Float16*)alloc((long)MP_E * 128 * 2);
    _Float16* nin  = (_Float16*)alloc((long)MP_N * 256 * 2);
    _Float16* tn1  = (_Float16*)alloc((long)MP_N * 128 * 2);
    _Float16* tn2  = (_Float16*)alloc((long)MP_N * 128 * 2);

    auto pack = [&](const float* W, int K, int N, int Kp, int Np) -> _Float16* {
        _Float16* Wt = (_Float16*)alloc((long)Np * Kp * 2);
        int total = Np * Kp;
        pack_wt_kernel<<<(total + 255) / 256, 256, 0, stream>>>(W, Wt, K, N, Kp, Np);
        return Wt;
    };

    _Float16 *Wen[3], *Wee[3], *Wdec[3], *WE[5][3], *WN[5][3];
    Wen[0] = pack(enW[0], 30, 128, 32, 128);
    Wen[1] = pack(enW[1], 128, 128, 128, 128);
    Wen[2] = pack(enW[2], 128, 128, 128, 128);
    Wee[0] = pack(eeW[0], 3, 128, 32, 128);
    Wee[1] = pack(eeW[1], 128, 128, 128, 128);
    Wee[2] = pack(eeW[2], 128, 128, 128, 128);
    Wdec[0] = pack(decW[0], 128, 128, 128, 128);
    Wdec[1] = pack(decW[1], 128, 128, 128, 128);
    Wdec[2] = pack(decW[2], 128, 3, 128, 16);     // N padded to one 16-tile
    for (int s = 0; s < 5; ++s) {
        WE[s][0] = pack(eW[s][0], 384, 128, 384, 128);
        WE[s][1] = pack(eW[s][1], 128, 128, 128, 128);
        WE[s][2] = pack(eW[s][2], 128, 128, 128, 128);
        WN[s][0] = pack(nW[s][0], 256, 128, 256, 128);
        WN[s][1] = pack(nW[s][1], 128, 128, 128, 128);
        WN[s][2] = pack(nW[s][2], 128, 128, 128, 128);
    }

    auto G = [&](const _Float16* A, int M, int Mpad, int K, const _Float16* Wt,
                 const float* b, const float* g, const float* bb,
                 float* resid, float* outf, _Float16* outh,
                 int nt, int Nout, int ldo, int fl) {
        gemm_wmma_kernel<<<Mpad / 64, 256, 0, stream>>>(
            A, Wt, b, g, bb, resid, outf, outh, M, K, nt, Nout, ldo, fl);
    };
    auto blk = [](long n) { return (int)((n + 255) / 256); };

    // Convert inputs (pad K to 32).
    cvt_pad_kernel<<<blk((long)N_NODES * 32), 256, 0, stream>>>(x, x16, N_NODES, 30, 32);
    cvt_pad_kernel<<<blk((long)N_EDGES * 32), 256, 0, stream>>>(efeat, ef16, N_EDGES, 3, 32);

    // Encoders: MLP(ReLU,ReLU,Identity) + LayerNorm.
    G(x16,  N_NODES, MP_N, 32,  Wen[0], enB[0], 0, 0, 0, 0, tn1, 8, 128, 128, F_RELU);
    G(tn1,  N_NODES, MP_N, 128, Wen[1], enB[1], 0, 0, 0, 0, tn2, 8, 128, 128, F_RELU);
    G(tn2,  N_NODES, MP_N, 128, Wen[2], enB[2], en_g, en_b, 0, hF, h16, 8, 128, 128, F_LN);
    G(ef16, N_EDGES, MP_E, 32,  Wee[0], eeB[0], 0, 0, 0, 0, te1, 8, 128, 128, F_RELU);
    G(te1,  N_EDGES, MP_E, 128, Wee[1], eeB[1], 0, 0, 0, 0, te2, 8, 128, 128, F_RELU);
    G(te2,  N_EDGES, MP_E, 128, Wee[2], eeB[2], ee_g, ee_b, 0, eF, e16, 8, 128, 128, F_LN);

    // Processor: 5 InteractionNetwork steps with residuals.
    for (int s = 0; s < 5; ++s) {
        gather_concat_kernel<<<blk((long)N_EDGES * 48), 256, 0, stream>>>(h16, e16, recv, send, msg);
        G(msg, N_EDGES, MP_E, 384, WE[s][0], eB[s][0], 0, 0, 0, 0, te1, 8, 128, 128, F_RELU);
        G(te1, N_EDGES, MP_E, 128, WE[s][1], eB[s][1], 0, 0, 0, 0, te2, 8, 128, 128, F_RELU);
        // e_new -> enew (for segment_sum); e += e_new (fp32 master + f16 mirror)
        G(te2, N_EDGES, MP_E, 128, WE[s][2], eB[s][2], elg[s], elb[s], eF, enew, e16, 8, 128, 128, F_LN);

        fill0_kernel<<<blk((long)N_NODES * 128), 256, 0, stream>>>(agg, (long)N_NODES * 128);
        scatter_add_kernel<<<blk((long)N_EDGES * 128), 256, 0, stream>>>(enew, recv, agg);
        concat_node_kernel<<<blk((long)N_NODES * 256), 256, 0, stream>>>(agg, h16, nin);

        G(nin, N_NODES, MP_N, 256, WN[s][0], nB[s][0], 0, 0, 0, 0, tn1, 8, 128, 128, F_RELU);
        G(tn1, N_NODES, MP_N, 128, WN[s][1], nB[s][1], 0, 0, 0, 0, tn2, 8, 128, 128, F_RELU);
        // h += h_new
        G(tn2, N_NODES, MP_N, 128, WN[s][2], nB[s][2], nlg[s], nlb[s], hF, 0, h16, 8, 128, 128, F_LN);
    }

    // Decoder -> d_out [10000 x 3] fp32.
    G(h16, N_NODES, MP_N, 128, Wdec[0], decB[0], 0, 0, 0, 0, tn1, 8, 128, 128, F_RELU);
    G(tn1, N_NODES, MP_N, 128, Wdec[1], decB[1], 0, 0, 0, 0, tn2, 8, 128, 128, F_RELU);
    G(tn2, N_NODES, MP_N, 128, Wdec[2], decB[2], 0, 0, 0, (float*)d_out, 0, 1, 3, 3, 0);
}
